// StructureSeqCrossAttn_43800076485386
// MI455X (gfx1250) — matrix operand order repaired and checked by reading, not compiled, not run on hardware
//
#include <hip/hip_runtime.h>
#include <math.h>

// ---------------------------------------------------------------------------
// StructureSeqCrossAttn for MI455X (gfx1250, wave32, WMMA bf16 16x16x32)
//  - all GEMMs on v_wmma_f32_16x16x32_bf16, f32 accumulate
//  - GEMM waves register-blocked 32x64 (8 WMMA per k-step, A/B frag reuse)
//  - attention: S tile staged in LDS, native v_sqrt/v_rcp for bias/softmax
// ---------------------------------------------------------------------------

typedef __attribute__((ext_vector_type(16))) __bf16 v16bf;
typedef __attribute__((ext_vector_type(8)))  __bf16 v8bf;
typedef __attribute__((ext_vector_type(8)))  float  v8f;

#define BB   8
#define GG   512
#define LLEN 1024
#define DM   448
#define PLM  1280
#define NH   7
#define DHD  64
#define TAU_F 3.0f

// ---- fragment builders (layouts per CDNA5 ISA 7.12.2) ----------
// A (16x32 bf16): lane l -> row m=l&15, khalf=l>>4; elems 0..7 = k0+8*kh+e,
// elems 8..15 = k0+16+8*kh+e  => two contiguous 16-byte loads.
__device__ __forceinline__ v16bf frag_a(const __bf16* __restrict__ p) {
  v8bf lo = *(const v8bf*)p;
  v8bf hi = *(const v8bf*)(p + 16);
  v16bf r;
#pragma unroll
  for (int e = 0; e < 8; ++e) { r[e] = lo[e]; r[e + 8] = hi[e]; }
  return r;
}
// B (32x16 bf16): lane l -> col n=l&15, khalf=l>>4; elems 0..7 = k0+8*kh+e,
// elems 8..15 = k0+16+8*kh+e  => two contiguous 16-byte loads.
__device__ __forceinline__ v16bf frag_b(const __bf16* __restrict__ p) {
  v8bf lo = *(const v8bf*)p;
  v8bf hi = *(const v8bf*)(p + 16);
  v16bf r;
#pragma unroll
  for (int e = 0; e < 8; ++e) { r[e] = lo[e]; r[e + 8] = hi[e]; }
  return r;
}

#define WMMA_BF16(Afrag, Bfrag, Cacc) \
  __builtin_amdgcn_wmma_f32_16x16x32_bf16(false, (Afrag), false, (Bfrag), (short)0, (Cacc), false, false)

// ---- fp32 -> bf16 conversion ----------------------------------------------
__global__ void cvt_f32_bf16(const float* __restrict__ s, __bf16* __restrict__ d, int n) {
  int i = blockIdx.x * blockDim.x + threadIdx.x;
  int stride = gridDim.x * blockDim.x;
  for (; i < n; i += stride) d[i] = (__bf16)s[i];
}

// ---- generic GEMM: C[M,N] = A[M,Kd] * W[N,Kd]^T  (bf16 in, f32 acc) -------
// Each wave computes a 32x64 tile: MB=2 row-tiles x NB=4 col-tiles of 16x16.
// Per k-step: 2 A-frag loads + 4 B-frag loads feed 8 WMMAs.
// OUT_KIND: 0 = f32 row-major, 1 = bf16 row-major, 2 = bf16 batch-transposed
//           (kind 2: C[b][n][m_local], ldc = rowsPerBatch)
template <int OUT_KIND>
__global__ __launch_bounds__(128) void gemm_bf16_xWt(
    const __bf16* __restrict__ A, int lda,
    const __bf16* __restrict__ W, int ldw,
    void* __restrict__ Cv, int ldc,
    int M, int N, int Kd, int rowsPerBatch) {
  const int lane = threadIdx.x & 31;
  const int warp = threadIdx.x >> 5;
  const int tilesN = N >> 6;                      // N/64 column super-tiles
  const int totalTiles = (M >> 5) * tilesN;       // M/32 row super-tiles
  const int tile = blockIdx.x * (blockDim.x >> 5) + warp;  // wave-uniform
  if (tile >= totalTiles) return;                 // uniform exit, EXEC stays full
  const int row0 = (tile / tilesN) << 5;
  const int n0   = (tile % tilesN) << 6;
  const int mloc = lane & 15;
  const int kh   = lane >> 4;

  const __bf16* aRow0 = A + (size_t)(row0 + mloc) * lda + kh * 8;
  const __bf16* aRow1 = aRow0 + (size_t)16 * lda;
  const __bf16* wRow  = W + (size_t)(n0 + mloc) * ldw + kh * 8;
  const size_t wTileStride = (size_t)16 * ldw;

  v8f acc[2][4] = {};
  for (int k0 = 0; k0 < Kd; k0 += 32) {
    const v16bf a0 = frag_a(aRow0 + k0);
    const v16bf a1 = frag_a(aRow1 + k0);
#pragma unroll
    for (int nt = 0; nt < 4; ++nt) {
      const v16bf bf = frag_b(wRow + (size_t)nt * wTileStride + k0);
      acc[0][nt] = WMMA_BF16(a0, bf, acc[0][nt]);
      acc[1][nt] = WMMA_BF16(a1, bf, acc[1][nt]);
    }
  }

  // C/D layout: VGPR r -> M = r + 8*kh, N = lane&15
#pragma unroll
  for (int mi = 0; mi < 2; ++mi) {
#pragma unroll
    for (int nt = 0; nt < 4; ++nt) {
#pragma unroll
      for (int r = 0; r < 8; ++r) {
        const int m = row0 + mi * 16 + r + 8 * kh;
        const int n = n0 + nt * 16 + mloc;
        const float v = acc[mi][nt][r];
        if constexpr (OUT_KIND == 0) {
          ((float*)Cv)[(size_t)m * ldc + n] = v;
        } else if constexpr (OUT_KIND == 1) {
          ((__bf16*)Cv)[(size_t)m * ldc + n] = (__bf16)v;
        } else {
          const int b  = m / rowsPerBatch;
          const int lr = m - b * rowsPerBatch;
          ((__bf16*)Cv)[((size_t)b * N + n) * (size_t)ldc + lr] = (__bf16)v;
        }
      }
    }
  }
}

// ---- fused attention: softmax(Q K^T/8 - dist/tau + mask) V  ----------------
// one wave handles 16 anchor rows of one (b, h). S tile staged in LDS (f32).
__global__ __launch_bounds__(32) void attn_kernel(
    const __bf16* __restrict__ Q,    // [B,G,D]  bf16
    const __bf16* __restrict__ K,    // [B,L,D]  bf16
    const __bf16* __restrict__ Vt,   // [B,D,L]  bf16 (transposed)
    const float*  __restrict__ axyz, // [B,G,3]
    const float*  __restrict__ sxyz, // [B,L,3]
    const unsigned char* __restrict__ mask, // [B,L]
    __bf16* __restrict__ ctx) {      // [B,G,D]  bf16
  extern __shared__ float Sbuf[];          // 16*LLEN f32 + 32 f32 stats
  float* stats = Sbuf + 16 * LLEN;         // [0..15]=rowmax, [16..31]=1/rowsum

  const int lane = threadIdx.x;
  const int g0 = blockIdx.x << 4;
  const int h  = blockIdx.y;
  const int b  = blockIdx.z;
  const int mloc = lane & 15;
  const int kh   = lane >> 4;

  // Q fragments for d = 0..31 and 32..63 (resident for whole kernel)
  const __bf16* qp = Q + ((size_t)b * GG + g0 + mloc) * DM + h * DHD + kh * 8;
  const v16bf qa0 = frag_a(qp);
  const v16bf qa1 = frag_a(qp + 32);

  // anchor xyz for the 8 rows this lane owns in C-fragments (m = r + 8*kh)
  float ax[8], ay[8], az[8];
#pragma unroll
  for (int r = 0; r < 8; ++r) {
    const float* p = axyz + ((size_t)b * GG + g0 + r + 8 * kh) * 3;
    ax[r] = p[0]; ay[r] = p[1]; az[r] = p[2];
  }

  // ---- pass 1: logits into LDS -------------------------------------------
  const float scale = 0.125f;  // 1/sqrt(64)
  for (int l0 = 0; l0 < LLEN; l0 += 16) {
    const __bf16* kp = K + ((size_t)b * LLEN + l0 + mloc) * DM + h * DHD + kh * 8;
    const v16bf kb0 = frag_b(kp);
    const v16bf kb1 = frag_b(kp + 32);
    v8f s = {};
    s = WMMA_BF16(qa0, kb0, s);
    s = WMMA_BF16(qa1, kb1, s);

    const int l = l0 + mloc;
    const float* sp = sxyz + ((size_t)b * LLEN + l) * 3;
    const float sx = sp[0], sy = sp[1], sz = sp[2];
    const bool ok = mask[(size_t)b * LLEN + l] != 0;
#pragma unroll
    for (int r = 0; r < 8; ++r) {
      const float dx = ax[r] - sx, dy = ay[r] - sy, dz = az[r] - sz;
      // native v_sqrt_f32 (TRANS op, co-executes with WMMA); bias term only
      const float dist = __builtin_amdgcn_sqrtf(dx * dx + dy * dy + dz * dz);
      float v = s[r] * scale - dist * (1.0f / TAU_F);
      if (!ok) v = -1.0e30f;
      Sbuf[(r + 8 * kh) * LLEN + l] = v;
    }
  }
  __syncthreads();  // single wave: LDS stays in-order; fence for the optimizer

  // ---- pass 2: row max, exp, row sum (wave-cooperative, shfl_xor) --------
  for (int m = 0; m < 16; ++m) {
    float v = -3.0e38f;
    for (int i = lane; i < LLEN; i += 32) v = fmaxf(v, Sbuf[m * LLEN + i]);
#pragma unroll
    for (int off = 16; off >= 1; off >>= 1) v = fmaxf(v, __shfl_xor(v, off, 32));
    if (lane == 0) stats[m] = v;
  }
  for (int m = 0; m < 16; ++m) {
    const float mx = stats[m];
    float sum = 0.0f;
    for (int i = lane; i < LLEN; i += 32) {
      const float p = __expf(Sbuf[m * LLEN + i] - mx);
      Sbuf[m * LLEN + i] = p;
      sum += p;
    }
#pragma unroll
    for (int off = 16; off >= 1; off >>= 1) sum += __shfl_xor(sum, off, 32);
    if (lane == 0) stats[16 + m] = __builtin_amdgcn_rcpf(sum);  // native v_rcp_f32
  }
  __syncthreads();

  // ---- pass 3: ctx = P * V  (K-loop over L, 4 N-tiles of 16 over d_head) --
  v8f acc[4] = {};
  for (int kc = 0; kc < LLEN; kc += 32) {
    const float* pr = Sbuf + mloc * LLEN + kc + kh * 8;
    v16bf pa;
#pragma unroll
    for (int e = 0; e < 8; ++e) { pa[e] = (__bf16)pr[e]; pa[e + 8] = (__bf16)pr[16 + e]; }
#pragma unroll
    for (int nt = 0; nt < 4; ++nt) {
      const __bf16* vp = Vt + ((size_t)b * DM + h * DHD + nt * 16 + mloc) * LLEN + kc + kh * 8;
      const v16bf vb = frag_b(vp);
      acc[nt] = WMMA_BF16(pa, vb, acc[nt]);
    }
  }

#pragma unroll
  for (int nt = 0; nt < 4; ++nt) {
#pragma unroll
    for (int r = 0; r < 8; ++r) {
      const int m = r + 8 * kh;
      const float inv = stats[16 + m];
      const int g = g0 + m;
      const int d = h * DHD + nt * 16 + mloc;
      ctx[((size_t)b * GG + g) * DM + d] = (__bf16)(acc[nt][r] * inv);
    }
  }
}

// ---------------------------------------------------------------------------
extern "C" void kernel_launch(void* const* d_in, const int* in_sizes, int n_in,
                              void* d_out, int out_size, void* d_ws, size_t ws_size,
                              hipStream_t stream) {
  const float* anchor_xyz  = (const float*)d_in[0];
  const float* anchor_feat = (const float*)d_in[1];
  const float* seq_xyz     = (const float*)d_in[2];
  const float* seq_esm     = (const float*)d_in[3];
  const unsigned char* seq_mask = (const unsigned char*)d_in[4];  // jnp bool -> 1 byte
  const float* Wq = (const float*)d_in[5];
  const float* Wk = (const float*)d_in[6];
  const float* Wv = (const float*)d_in[7];
  const float* Wo = (const float*)d_in[8];
  float* out = (float*)d_out;
  (void)in_sizes; (void)n_in; (void)out_size; (void)ws_size;

  char* ws = (char*)d_ws;
  size_t off = 0;
  auto carve = [&](size_t elems) -> __bf16* {
    __bf16* p = (__bf16*)(ws + off);
    off += (elems * sizeof(__bf16) + 255) & ~(size_t)255;
    return p;
  };
  __bf16* abf  = carve((size_t)BB * GG * DM);      // anchor_feat bf16
  __bf16* ebf  = carve((size_t)BB * LLEN * PLM);   // seq_esm bf16
  __bf16* wqb  = carve((size_t)DM * DM);
  __bf16* wkb  = carve((size_t)DM * PLM);
  __bf16* wvb  = carve((size_t)DM * PLM);
  __bf16* wob  = carve((size_t)DM * DM);
  __bf16* Qb   = carve((size_t)BB * GG * DM);      // [B,G,D]
  __bf16* Kb   = carve((size_t)BB * LLEN * DM);    // [B,L,D]
  __bf16* Vtb  = carve((size_t)BB * DM * LLEN);    // [B,D,L] transposed
  __bf16* ctxb = carve((size_t)BB * GG * DM);      // [B,G,D]

  auto cvt = [&](const float* s, __bf16* d, size_t n) {
    int blocks = (int)((n + 255) / 256);
    if (blocks > 2048) blocks = 2048;
    cvt_f32_bf16<<<blocks, 256, 0, stream>>>(s, d, (int)n);
  };
  cvt(anchor_feat, abf, (size_t)BB * GG * DM);
  cvt(seq_esm,     ebf, (size_t)BB * LLEN * PLM);
  cvt(Wq, wqb, (size_t)DM * DM);
  cvt(Wk, wkb, (size_t)DM * PLM);
  cvt(Wv, wvb, (size_t)DM * PLM);
  cvt(Wo, wob, (size_t)DM * DM);

  // Q = anchor_feat @ Wq^T      [4096,448] x [448,448]^T
  {
    const int M = BB * GG, N = DM, Kd = DM;
    const int tiles = (M / 32) * (N / 64);
    gemm_bf16_xWt<1><<<(tiles + 3) / 4, 128, 0, stream>>>(abf, DM, wqb, DM, Qb, DM, M, N, Kd, 1);
  }
  // K = seq_esm @ Wk^T          [8192,1280] x [448,1280]^T
  {
    const int M = BB * LLEN, N = DM, Kd = PLM;
    const int tiles = (M / 32) * (N / 64);
    gemm_bf16_xWt<1><<<(tiles + 3) / 4, 128, 0, stream>>>(ebf, PLM, wkb, PLM, Kb, DM, M, N, Kd, 1);
  }
  // V^T (stored [B, D, L]) for contiguous P*V B-fragments
  {
    const int M = BB * LLEN, N = DM, Kd = PLM;
    const int tiles = (M / 32) * (N / 64);
    gemm_bf16_xWt<2><<<(tiles + 3) / 4, 128, 0, stream>>>(ebf, PLM, wvb, PLM, Vtb, LLEN, M, N, Kd, LLEN);
  }
  // fused attention: grid (G/16, H, B), one wave per block, 64KB+ LDS
  {
    const size_t smem = (size_t)(16 * LLEN + 32) * sizeof(float);  // 65664 B < 320KB/WGP
    hipFuncSetAttribute((const void*)attn_kernel,
                        hipFuncAttributeMaxDynamicSharedMemorySize, (int)smem);
    attn_kernel<<<dim3(GG / 16, NH, BB), 32, smem, stream>>>(
        Qb, Kb, Vtb, anchor_xyz, seq_xyz, seq_mask, ctxb);
  }
  // out = ctx @ Wo^T (f32 output)
  {
    const int M = BB * GG, N = DM, Kd = DM;
    const int tiles = (M / 32) * (N / 64);
    gemm_bf16_xWt<0><<<(tiles + 3) / 4, 128, 0, stream>>>(ctxb, DM, wob, DM, out, DM, M, N, Kd, 1);
  }
}
